// SparseCoordinateHead_11759620456695
// MI455X (gfx1250) — matrix operand order repaired and checked by reading, not compiled, not run on hardware
//
#include <hip/hip_runtime.h>
#include <hip/hip_bf16.h>
#include <math.h>

// ---------------------------------------------------------------------------
// SparseCoordinateHead on MI455X (gfx1250, wave32, WMMA)
//
// Edge MLP  feat[E,257] @ W1^T[257,128] -> SiLU -> @W2[128] -> tanh, mapped to
// v_wmma_f32_16x16x32_f16 (f32 accumulate):
//   - one wave = 16 edges (one M-tile); 8 N-tiles; 8 K-steps of 32
//     (K 0..127 from h[row], 128..255 from h[col]); the d^2*W1[:,256]+b1
//     rank-1 term is folded into the C initializer.
//   - W1 staged once per block into LDS as f16 (64KB of 320KB/WGP) in
//     B-fragment order: each B operand is 2x ds_read_b128 right before use.
//   - kt loop kept rolled (#pragma unroll 1): one A-fragment + in-flight
//     B-fragments live next to the 64 accumulator VGPRs -> no scratch spills.
//   - Epilogue uses raw v_rcp_f32 / v_rsq_f32 (rcpf/rsqf builtins) instead of
//     the IEEE div/sqrt expansions: SiLU and tanh are on learned weights and
//     tolerate 1-ulp-ish transcendentals; this keeps the VALU pipe free to
//     co-execute under other waves' XDL WMMA work.
// ---------------------------------------------------------------------------

typedef __attribute__((ext_vector_type(16))) _Float16 v16h;
typedef __attribute__((ext_vector_type(2)))  __fp16   fp16x2;  // cvt_pkrtz result
typedef __attribute__((ext_vector_type(8)))  float    v8f;
typedef __attribute__((ext_vector_type(4)))  float    f4;

#define DIM    128
#define FANIN  257          // 2*DIM + 1
#define TILE_E 16
#define WAVES_PER_BLOCK 8
#define THREADS (WAVES_PER_BLOCK * 32)

__global__ void __launch_bounds__(256)
zero_out_kernel(float* __restrict__ out, int n) {
    int i = blockIdx.x * blockDim.x + threadIdx.x;
    if (i < n) out[i] = 0.0f;
}

union AFrag {
    v16h   v;
    fp16x2 p[8];
};

__global__ void __launch_bounds__(THREADS)
edge_mlp_wmma_kernel(const float* __restrict__ h,
                     const float* __restrict__ x,
                     const int*   __restrict__ ei,   // [2, E] int32
                     const float* __restrict__ W1,   // [128, 257] row-major
                     const float* __restrict__ b1,   // [128]
                     const float* __restrict__ W2,   // [128]
                     float* __restrict__ out,        // [N_NODES, 2]
                     int nEdges, int nTiles)
{
    // W1 (k < 256) as f16, [n][k] row-major.
    __shared__ __align__(32) _Float16 sW1[DIM * 256];
    for (int i = threadIdx.x; i < DIM * 256; i += THREADS) {
        int n = i >> 8, k = i & 255;
        sW1[i] = (_Float16)W1[n * FANIN + k];
    }
    __syncthreads();

    const int lane = threadIdx.x & 31;
    const int wave = threadIdx.x >> 5;
    const int l = lane & 15;        // N column within a 16-wide N-tile
    const int g = lane >> 4;        // half-wave group (WMMA M/K split)

    // Per-lane constants for output column l of each N-tile.
    float b1v[8], wlastv[8], w2v[8];
#pragma unroll
    for (int t = 0; t < 8; ++t) {
        int n = t * 16 + l;
        b1v[t]    = b1[n];
        wlastv[t] = W1[n * FANIN + 256];  // d^2 column of W1
        w2v[t]    = W2[n];
    }

    const int waveId     = blockIdx.x * WAVES_PER_BLOCK + wave;
    const int waveStride = gridDim.x * WAVES_PER_BLOCK;

    // Base of this lane's B-fragment column in LDS (row = t*16 + l).
    const _Float16* sW1lane = &sW1[(size_t)l * 256 + g * 16];

#pragma unroll 1
    for (int tile = waveId; tile < nTiles; tile += waveStride) {
        // Lanes 16-31 mirror lanes 0-15: every lane owns edge (tile*16 + l).
        int e = tile * TILE_E + l;
        if (e >= nEdges) e = nEdges - 1;
        const int er = ei[e];
        const int ec = ei[nEdges + e];
        const float rx  = x[2 * er]     - x[2 * ec];
        const float ry  = x[2 * er + 1] - x[2 * ec + 1];
        const float dsq = rx * rx + ry * ry;

        // C init: acc[t][r] = d^2[m]*W1[n,256] + b1[n],  m = r + 8g, n = 16t + l
        v8f acc[8];
#pragma unroll
        for (int r = 0; r < 8; ++r) {
            float dm = __shfl(dsq, r + 8 * g, 32);
#pragma unroll
            for (int t = 0; t < 8; ++t)
                acc[t][r] = dm * wlastv[t] + b1v[t];
        }

        const float* __restrict__ rowR = h + (size_t)er * DIM;
        const float* __restrict__ rowC = h + (size_t)ec * DIM;

        // K loop: 8 steps of 32. Rolled on purpose (see header comment).
#pragma unroll 1
        for (int kt = 0; kt < 8; ++kt) {
            const float* src = (kt < 4) ? rowR : rowC;
            const int koff = (kt & 3) * 32 + g * 16;  // contiguous A layout, group g
            // 16-bit A 16x32: lane holds K = koff..koff+15 (matches B)
            f4 a0 = *(const f4*)(src + koff);
            f4 a1 = *(const f4*)(src + koff + 4);
            f4 a2 = *(const f4*)(src + koff + 8);
            f4 a3 = *(const f4*)(src + koff + 12);
            AFrag A;
            A.p[0] = __builtin_amdgcn_cvt_pkrtz(a0[0], a0[1]);
            A.p[1] = __builtin_amdgcn_cvt_pkrtz(a0[2], a0[3]);
            A.p[2] = __builtin_amdgcn_cvt_pkrtz(a1[0], a1[1]);
            A.p[3] = __builtin_amdgcn_cvt_pkrtz(a1[2], a1[3]);
            A.p[4] = __builtin_amdgcn_cvt_pkrtz(a2[0], a2[1]);
            A.p[5] = __builtin_amdgcn_cvt_pkrtz(a2[2], a2[3]);
            A.p[6] = __builtin_amdgcn_cvt_pkrtz(a3[0], a3[1]);
            A.p[7] = __builtin_amdgcn_cvt_pkrtz(a3[2], a3[3]);

            const _Float16* bk = sW1lane + kt * 32;
#pragma unroll
            for (int t = 0; t < 8; ++t) {
                // B 32x16: lane = column l, group g holds K = g*16..g*16+15,
                // 16 contiguous halves -> 2x ds_read_b128 right before use.
                v16h B = *(const v16h*)(bk + (size_t)t * 16 * 256);
                acc[t] = __builtin_amdgcn_wmma_f32_16x16x32_f16(
                    false, A.v, false, B, (short)0, acc[t], false, false);
            }
        }

        // Epilogue: SiLU, dot with W2, reduce over the 16 N-lanes per group.
        float p[8];
#pragma unroll
        for (int r = 0; r < 8; ++r) p[r] = 0.0f;
#pragma unroll
        for (int t = 0; t < 8; ++t) {
#pragma unroll
            for (int r = 0; r < 8; ++r) {
                float v   = acc[t][r];
                // SiLU via raw v_rcp_f32 (no IEEE div expansion)
                float hid = v * __builtin_amdgcn_rcpf(1.0f + __expf(-v));
                p[r] += hid * w2v[t];
            }
        }
#pragma unroll
        for (int r = 0; r < 8; ++r) {
            p[r] += __shfl_xor(p[r], 1, 32);
            p[r] += __shfl_xor(p[r], 2, 32);
            p[r] += __shfl_xor(p[r], 4, 32);
            p[r] += __shfl_xor(p[r], 8, 32);   // 16-lane group sum
        }

        // Edge m = g*8 + (l&7) finalized by lanes with l < 8 in each group.
        const int   m   = g * 8 + (l & 7);
        const float s   = p[l & 7];
        // tanh(s/10) = (e^{2s/10}-1) / (e^{2s/10}+1), rcp-based
        const float e2  = __expf(s * 0.2f);
        const float w   = (e2 - 1.0f) * __builtin_amdgcn_rcpf(e2 + 1.0f);
        const float rxm = __shfl(rx,  m, 32);
        const float rym = __shfl(ry,  m, 32);
        const float dqm = __shfl(dsq, m, 32);
        const int   rim = __shfl(er,  m, 32);
        if ((l & 15) < 8 && (tile * TILE_E + m) < nEdges) {
            // w / (dij * K)
            const float scale = w * __builtin_amdgcn_rsqf(dqm + 1e-8f) * (1.0f / 32.0f);
            atomicAdd(&out[2 * rim],     rxm * scale);
            atomicAdd(&out[2 * rim + 1], rym * scale);
        }
    }
}

extern "C" void kernel_launch(void* const* d_in, const int* in_sizes, int n_in,
                              void* d_out, int out_size, void* d_ws, size_t ws_size,
                              hipStream_t stream) {
    const float* h  = (const float*)d_in[0];
    const float* x  = (const float*)d_in[1];
    const int*   ei = (const int*)  d_in[2];   // JAX (x64 off) -> int32
    const float* W1 = (const float*)d_in[3];
    const float* b1 = (const float*)d_in[4];
    const float* W2 = (const float*)d_in[5];
    float* out = (float*)d_out;

    const int nEdges = in_sizes[2] / 2;
    const int nTiles = (nEdges + TILE_E - 1) / TILE_E;

    zero_out_kernel<<<(out_size + 255) / 256, 256, 0, stream>>>(out, out_size);

    int blocks = (nTiles + WAVES_PER_BLOCK - 1) / WAVES_PER_BLOCK;
    if (blocks > 1280) blocks = 1280;   // grid-stride; amortize 64KB LDS staging
    edge_mlp_wmma_kernel<<<blocks, THREADS, 0, stream>>>(
        h, x, ei, W1, b1, W2, out, nEdges, nTiles);
}
